// TypeDP_26156350833071
// MI455X (gfx1250) — compile-verified, matrix operand order checked by reading, not run on hardware
//
#include <hip/hip_runtime.h>
#include <hip/hip_bf16.h>

typedef __attribute__((ext_vector_type(16))) _Float16 v16h;
typedef __attribute__((ext_vector_type(8)))  float    v8f;

#define TPB   256
#define WAVES 8
#define KNBR  200
#define KPAD  224
#define MR    208   // 13 M-tiles of 16 covering 200 rows

// ---- branch-free tanh: v_exp_f32 + v_rcp_f32, co-executes with WMMA -------
__device__ __forceinline__ float fast_tanh(float x) {
    // tanh(x) = 1 - 2/(exp(2x)+1); exp(2x) = exp2(x * 2*log2(e))
    float e = __builtin_amdgcn_exp2f(x * 2.8853900817779268f);
    return 1.0f - 2.0f * __builtin_amdgcn_rcpf(e + 1.0f);
}

// ---- WMMA fragment helpers (gfx1250 wave32, f16 A/B, f32 C/D) -------------
// 16-bit A/B fragment: element e of v16h holds K = (e&7) + (e>>3)*16 + (lane>>4)*8
// => two contiguous 8-half (16B) chunks at offsets k0+hi8 and k0+hi8+16.
__device__ __forceinline__ v16h frag16(const _Float16* p) {
    union { v16h v; uint4 q[2]; } u;
    u.q[0] = *(const uint4*)(p);
    u.q[1] = *(const uint4*)(p + 16);
    return u.v;
}

__device__ __forceinline__ v8f wmma_f16(v16h a, v16h b, v8f c) {
    return __builtin_amdgcn_wmma_f32_16x16x32_f16(false, a, false, b,
                                                  (short)0, c, false, false);
}

// ---------------------------------------------------------------------------
// Prep 1: type-embedding MLP folded into c0; small weights transposed to f16.
// ---------------------------------------------------------------------------
__global__ __launch_bounds__(TPB) void prep_small(
    const float* __restrict__ tv,  const float* __restrict__ tw0,
    const float* __restrict__ tb0, const float* __restrict__ tw1,
    const float* __restrict__ tb1, const float* __restrict__ ew0,
    const float* __restrict__ eb0, const float* __restrict__ ew1,
    const float* __restrict__ eb1, const float* __restrict__ ew2,
    const float* __restrict__ eb2,
    const float* __restrict__ fb0, const float* __restrict__ fb1,
    const float* __restrict__ fb2, const float* __restrict__ fw3,
    const float* __restrict__ fb3,
    float* __restrict__ c0,   float* __restrict__ w0,
    float* __restrict__ b1p,  float* __restrict__ b2p,
    float* __restrict__ fb0p, float* __restrict__ fb1p,
    float* __restrict__ fb2p, float* __restrict__ fw3p,
    float* __restrict__ fb3c,
    _Float16* __restrict__ W1t, _Float16* __restrict__ W2t)
{
    __shared__ float sT1[200 * 8];
    __shared__ float sT2[200 * 8];
    const int tid = threadIdx.x;

    for (int i = tid; i < 1600; i += TPB) {
        int k = i >> 3, j = i & 7;
        float a = tb0[j];
        for (int p = 0; p < 4; ++p) a += tv[(k / 100) * 4 + p] * tw0[p * 8 + j];
        sT1[i] = fast_tanh(a);
    }
    __syncthreads();
    for (int i = tid; i < 1600; i += TPB) {
        int k = i >> 3, j = i & 7;
        float a = tb1[j];
        for (int p = 0; p < 8; ++p) a += sT1[k * 8 + p] * tw1[p * 8 + j];
        sT2[i] = fast_tanh(a);
    }
    __syncthreads();
    // c0[k][j] = t[k] @ ew0[1:9,:] + eb0   (layer-0 constant part)
    for (int i = tid; i < 200 * 25; i += TPB) {
        int k = i / 25, j = i % 25;
        float a = eb0[j];
        for (int p = 0; p < 8; ++p) a += sT2[k * 8 + p] * ew0[(1 + p) * 25 + j];
        c0[i] = a;
    }
    for (int i = tid; i < 32;  i += TPB) w0[i]  = (i < 25)  ? ew0[i] : 0.f;
    for (int i = tid; i < 64;  i += TPB) b1p[i] = (i < 50)  ? eb1[i] : 0.f;
    for (int i = tid; i < 112; i += TPB) b2p[i] = (i < 100) ? eb2[i] : 0.f;
    for (int i = tid; i < 512; i += TPB) {
        int nt = i >> 8, j = i & 255;
        fb0p[i] = (j < 240) ? fb0[nt * 240 + j] : 0.f;
        fb1p[i] = (j < 240) ? fb1[nt * 240 + j] : 0.f;
        fb2p[i] = (j < 240) ? fb2[nt * 240 + j] : 0.f;
        fw3p[i] = (j < 240) ? fw3[nt * 240 + j] : 0.f;
    }
    if (tid < 2) fb3c[tid] = fb3[tid];
    // W1t[n][k] = ew1[k][n], padded 50->64, 25->32 (zero pads)
    for (int i = tid; i < 64 * 32; i += TPB) {
        int n = i >> 5, k = i & 31;
        W1t[i] = (n < 50 && k < 25) ? (_Float16)ew1[k * 50 + n] : (_Float16)0.f;
    }
    // W2t[n][k] = ew2[k][n], padded 100->112, 50->64
    for (int i = tid; i < 112 * 64; i += TPB) {
        int n = i >> 6, k = i & 63;
        W2t[i] = (n < 100 && k < 50) ? (_Float16)ew2[k * 100 + n] : (_Float16)0.f;
    }
}

// ---------------------------------------------------------------------------
// Prep 2: fitting weights -> transposed padded f16 [nt][256][K]
// ---------------------------------------------------------------------------
__global__ __launch_bounds__(TPB) void prep_fit(
    const float* __restrict__ fw0, const float* __restrict__ fw1,
    const float* __restrict__ fw2,
    _Float16* __restrict__ F0t, _Float16* __restrict__ F1t,
    _Float16* __restrict__ F2t)
{
    const int NF0 = 2 * 256 * 1600, NF1 = 2 * 256 * 256;
    const int total = NF0 + 2 * NF1;
    for (int idx = blockIdx.x * TPB + threadIdx.x; idx < total;
         idx += gridDim.x * TPB) {
        if (idx < NF0) {
            int nt = idx / (256 * 1600), r = idx % (256 * 1600);
            int n = r / 1600, k = r % 1600;
            F0t[idx] = (n < 240) ? (_Float16)fw0[((size_t)nt * 1600 + k) * 240 + n]
                                 : (_Float16)0.f;
        } else if (idx < NF0 + NF1) {
            int j = idx - NF0;
            int nt = j / (256 * 256), r = j % (256 * 256);
            int n = r / 256, k = r % 256;
            F1t[j] = (n < 240 && k < 240)
                         ? (_Float16)fw1[((size_t)nt * 240 + k) * 240 + n]
                         : (_Float16)0.f;
        } else {
            int j = idx - NF0 - NF1;
            int nt = j / (256 * 256), r = j % (256 * 256);
            int n = r / 256, k = r % 256;
            F2t[j] = (n < 240 && k < 240)
                         ? (_Float16)fw2[((size_t)nt * 240 + k) * 240 + n]
                         : (_Float16)0.f;
        }
    }
}

// ---------------------------------------------------------------------------
// Kernel A: per-atom embedding net + xyz + dr -> H[atom][1600] (f16)
// ---------------------------------------------------------------------------
__global__ __launch_bounds__(TPB) void embed_kernel(
    const float* __restrict__ Ri, const float* __restrict__ c0,
    const float* __restrict__ w0, const float* __restrict__ b1p,
    const float* __restrict__ b2p, const _Float16* __restrict__ W1t,
    const _Float16* __restrict__ W2t, _Float16* __restrict__ H)
{
    __shared__ __align__(16) _Float16 sBlkT[16][KPAD];   // A of xyz: blk^T, K-pads zeroed
    __shared__ __align__(16) _Float16 sG0[MR][32];       // layer0 out (A of GEMM1)
    __shared__ __align__(16) _Float16 sG1[MR][64];       // layer1 out (A of GEMM2)
    __shared__ __align__(16) _Float16 sG2t[112][KPAD];   // layer2 out, transposed (B of xyz)
    __shared__ float sS[KNBR];
    __shared__ float sXyz[4][112];

    const int tid  = threadIdx.x;
    const int wave = tid >> 5, lane = tid & 31;
    const int ln   = lane & 15, hi8 = (lane >> 4) << 3;
    const size_t w = blockIdx.x;  // b*2048 + nt*1024 + atom

    // stage blk^T (f16) + s column; zero all pad cells (disjoint addresses)
    for (int i = tid; i < 16 * KPAD; i += TPB) {
        int c = i / KPAD, k = i % KPAD;
        if (c >= 4 || k >= KNBR) sBlkT[c][k] = (_Float16)0.f;
    }
    for (int k = tid; k < KNBR; k += TPB) {
        const float4 v = *(const float4*)(Ri + (w * KNBR + k) * 4);
        sS[k] = v.x;
        sBlkT[0][k] = (_Float16)v.x;
        sBlkT[1][k] = (_Float16)v.y;
        sBlkT[2][k] = (_Float16)v.z;
        sBlkT[3][k] = (_Float16)v.w;
    }
    __syncthreads();

    // layer 0: rank-1 + precomputed constant, tanh (VALU/TRANS)
    for (int i = tid; i < MR * 32; i += TPB) {
        int k = i >> 5, j = i & 31;
        float v = 0.f;
        if (k < KNBR && j < 25) v = fast_tanh(sS[k] * w0[j] + c0[k * 25 + j]);
        sG0[k][j] = (_Float16)v;
    }
    __syncthreads();

    // GEMM1: (208x32) @ (32x64) -> tanh -> sG1
    for (int job = wave; job < 13 * 4; job += WAVES) {
        int tm = job >> 2, tn = job & 3;
        v8f acc = {};
        v16h a = frag16(&sG0[tm * 16 + ln][hi8]);
        v16h b = frag16(&W1t[(size_t)(tn * 16 + ln) * 32 + hi8]);
        acc = wmma_f16(a, b, acc);
        float bias = b1p[tn * 16 + ln];
        #pragma unroll
        for (int r = 0; r < 8; ++r)
            sG1[tm * 16 + hi8 + r][tn * 16 + ln] = (_Float16)fast_tanh(acc[r] + bias);
    }
    __syncthreads();

    // GEMM2: (208x64) @ (64x112) -> tanh -> sG2t (transposed store)
    for (int job = wave; job < 13 * 7; job += WAVES) {
        int tm = job / 7, tn = job % 7;
        v8f acc = {};
        #pragma unroll
        for (int ks = 0; ks < 2; ++ks) {
            v16h a = frag16(&sG1[tm * 16 + ln][ks * 32 + hi8]);
            v16h b = frag16(&W2t[(size_t)(tn * 16 + ln) * 64 + ks * 32 + hi8]);
            acc = wmma_f16(a, b, acc);
        }
        float bias = b2p[tn * 16 + ln];
        #pragma unroll
        for (int r = 0; r < 8; ++r)
            sG2t[tn * 16 + ln][tm * 16 + hi8 + r] = (_Float16)fast_tanh(acc[r] + bias);
    }
    __syncthreads();

    // xyz = blk^T @ g / 200 : (16x224) @ (224x112), keep rows c<4
    for (int tn = wave; tn < 7; tn += WAVES) {
        v8f acc = {};
        #pragma unroll
        for (int ks = 0; ks < 7; ++ks) {
            v16h a = frag16(&sBlkT[ln][ks * 32 + hi8]);
            v16h b = frag16(&sG2t[tn * 16 + ln][ks * 32 + hi8]);
            acc = wmma_f16(a, b, acc);
        }
        #pragma unroll
        for (int r = 0; r < 8; ++r) {
            int m = hi8 + r;
            if (m < 4) sXyz[m][tn * 16 + ln] = acc[r] * (1.0f / 200.0f);
        }
    }
    __syncthreads();

    // dr[d][e] = sum_c xyz[c][d]*xyz[c][e] -> H row (f16)
    for (int i = tid; i < 1600; i += TPB) {
        int d = i >> 4, e = i & 15;
        float v = sXyz[0][d] * sXyz[0][e] + sXyz[1][d] * sXyz[1][e]
                + sXyz[2][d] * sXyz[2][e] + sXyz[3][d] * sXyz[3][e];
        H[w * 1600 + i] = (_Float16)v;
    }
}

// ---------------------------------------------------------------------------
// Kernel B: fitting net, 16 rows per WG, WMMA for all three 240-wide layers.
// ---------------------------------------------------------------------------
__global__ __launch_bounds__(TPB) void fit_kernel(
    const _Float16* __restrict__ H,   const _Float16* __restrict__ F0t,
    const _Float16* __restrict__ F1t, const _Float16* __restrict__ F2t,
    const float* __restrict__ fb0p, const float* __restrict__ fb1p,
    const float* __restrict__ fb2p, const float* __restrict__ fw3p,
    const float* __restrict__ fb3c, float* __restrict__ out)
{
    __shared__ __align__(16) _Float16 sH[16][800];
    __shared__ __align__(16) _Float16 sA[16][256];
    __shared__ __align__(16) _Float16 sB[16][256];

    const int tid  = threadIdx.x;
    const int wave = tid >> 5, lane = tid & 31;
    const int ln   = lane & 15, hi8 = (lane >> 4) << 3;
    const int nt   = blockIdx.x >> 8;
    const int g0   = (blockIdx.x & 255) * 16;                 // row base within type
    const size_t hrow0 = (size_t)(g0 >> 10) * 2048 + (size_t)nt * 1024 + (g0 & 1023);
    const int tn0  = wave * 2;                                // 2 N-tiles per wave

    // ---- layer 0: (16x1600) @ (1600x256), K staged in two 800-chunks ----
    v8f acc0[2] = {{}, {}};
    for (int c = 0; c < 2; ++c) {
        for (int i = tid; i < 16 * 100; i += TPB) {           // 100 uint4 per row
            int r = i / 100, q = i % 100;
            ((uint4*)&sH[r][0])[q] =
                *((const uint4*)(H + (hrow0 + r) * 1600 + c * 800) + q);
        }
        __syncthreads();
        for (int ks = 0; ks < 25; ++ks) {
            v16h a = frag16(&sH[ln][ks * 32 + hi8]);
            #pragma unroll
            for (int t = 0; t < 2; ++t) {
                v16h b = frag16(F0t + ((size_t)nt * 256 + (tn0 + t) * 16 + ln) * 1600
                                    + c * 800 + ks * 32 + hi8);
                acc0[t] = wmma_f16(a, b, acc0[t]);
            }
        }
        __syncthreads();
    }
    #pragma unroll
    for (int t = 0; t < 2; ++t) {
        int n = (tn0 + t) * 16 + ln;
        float bias = fb0p[nt * 256 + n];
        #pragma unroll
        for (int r = 0; r < 8; ++r)
            sA[hi8 + r][n] = (_Float16)fast_tanh(acc0[t][r] + bias);
    }
    __syncthreads();

    // ---- layer 1: (16x256) @ (256x256) -> sB ----
    v8f acc1[2] = {{}, {}};
    for (int ks = 0; ks < 8; ++ks) {
        v16h a = frag16(&sA[ln][ks * 32 + hi8]);
        #pragma unroll
        for (int t = 0; t < 2; ++t) {
            v16h b = frag16(F1t + ((size_t)nt * 256 + (tn0 + t) * 16 + ln) * 256
                                + ks * 32 + hi8);
            acc1[t] = wmma_f16(a, b, acc1[t]);
        }
    }
    #pragma unroll
    for (int t = 0; t < 2; ++t) {
        int n = (tn0 + t) * 16 + ln;
        float bias = fb1p[nt * 256 + n];
        #pragma unroll
        for (int r = 0; r < 8; ++r)
            sB[hi8 + r][n] = (_Float16)fast_tanh(acc1[t][r] + bias);
    }
    __syncthreads();

    // ---- layer 2: (16x256) @ (256x256) -> sA ----
    v8f acc2[2] = {{}, {}};
    for (int ks = 0; ks < 8; ++ks) {
        v16h a = frag16(&sB[ln][ks * 32 + hi8]);
        #pragma unroll
        for (int t = 0; t < 2; ++t) {
            v16h b = frag16(F2t + ((size_t)nt * 256 + (tn0 + t) * 16 + ln) * 256
                                + ks * 32 + hi8);
            acc2[t] = wmma_f16(a, b, acc2[t]);
        }
    }
    #pragma unroll
    for (int t = 0; t < 2; ++t) {
        int n = (tn0 + t) * 16 + ln;
        float bias = fb2p[nt * 256 + n];
        #pragma unroll
        for (int r = 0; r < 8; ++r)
            sA[hi8 + r][n] = (_Float16)fast_tanh(acc2[t][r] + bias);
    }
    __syncthreads();

    // ---- layer 3: per-row 240-dot + bias -> Ei ----
    if (tid < 16) {
        float ei = fb3c[nt];
        for (int k = 0; k < 240; ++k)
            ei += (float)sA[tid][k] * fw3p[nt * 256 + k];
        out[4 + hrow0 + tid] = ei;
    }
}

// ---------------------------------------------------------------------------
// Kernel C: Etot[b] = sum_i Ei[b][i]  (deterministic tree reduction)
// ---------------------------------------------------------------------------
__global__ __launch_bounds__(TPB) void etot_kernel(const float* __restrict__ Ei,
                                                   float* __restrict__ out)
{
    __shared__ float red[TPB];
    const int tid = threadIdx.x, b = blockIdx.x;
    float s = 0.f;
    for (int i = tid; i < 2048; i += TPB) s += Ei[(size_t)b * 2048 + i];
    red[tid] = s;
    __syncthreads();
    for (int o = 128; o > 0; o >>= 1) {
        if (tid < o) red[tid] += red[tid + o];
        __syncthreads();
    }
    if (tid == 0) out[b] = red[0];
}

// ---------------------------------------------------------------------------
extern "C" void kernel_launch(void* const* d_in, const int* in_sizes, int n_in,
                              void* d_out, int out_size, void* d_ws, size_t ws_size,
                              hipStream_t stream)
{
    const float* Ri  = (const float*)d_in[0];
    const float* tw0 = (const float*)d_in[1];
    const float* tb0 = (const float*)d_in[2];
    const float* tw1 = (const float*)d_in[3];
    const float* tb1 = (const float*)d_in[4];
    const float* ew0 = (const float*)d_in[5];
    const float* eb0 = (const float*)d_in[6];
    const float* ew1 = (const float*)d_in[7];
    const float* eb1 = (const float*)d_in[8];
    const float* ew2 = (const float*)d_in[9];
    const float* eb2 = (const float*)d_in[10];
    const float* fw0 = (const float*)d_in[11];
    const float* fb0 = (const float*)d_in[12];
    const float* fw1 = (const float*)d_in[13];
    const float* fb1 = (const float*)d_in[14];
    const float* fw2 = (const float*)d_in[15];
    const float* fb2 = (const float*)d_in[16];
    const float* fw3 = (const float*)d_in[17];
    const float* fb3 = (const float*)d_in[18];
    const float* tv  = (const float*)d_in[19];
    (void)in_sizes; (void)n_in; (void)out_size; (void)ws_size;

    char* ws = (char*)d_ws;
    size_t off = 0;
    auto alloc = [&](size_t bytes) -> void* {
        void* p = ws + off;
        off += (bytes + 255) & ~(size_t)255;
        return p;
    };
    float*    c0   = (float*)alloc(200 * 25 * 4);
    float*    w0   = (float*)alloc(32 * 4);
    float*    b1p  = (float*)alloc(64 * 4);
    float*    b2p  = (float*)alloc(112 * 4);
    float*    fb0p = (float*)alloc(512 * 4);
    float*    fb1p = (float*)alloc(512 * 4);
    float*    fb2p = (float*)alloc(512 * 4);
    float*    fw3p = (float*)alloc(512 * 4);
    float*    fb3c = (float*)alloc(2 * 4);
    _Float16* W1t  = (_Float16*)alloc(64 * 32 * 2);
    _Float16* W2t  = (_Float16*)alloc(112 * 64 * 2);
    _Float16* F0t  = (_Float16*)alloc((size_t)2 * 256 * 1600 * 2);
    _Float16* F1t  = (_Float16*)alloc((size_t)2 * 256 * 256 * 2);
    _Float16* F2t  = (_Float16*)alloc((size_t)2 * 256 * 256 * 2);
    _Float16* H    = (_Float16*)alloc((size_t)8192 * 1600 * 2);

    prep_small<<<1, TPB, 0, stream>>>(tv, tw0, tb0, tw1, tb1, ew0, eb0, ew1, eb1,
                                      ew2, eb2, fb0, fb1, fb2, fw3, fb3,
                                      c0, w0, b1p, b2p, fb0p, fb1p, fb2p, fw3p,
                                      fb3c, W1t, W2t);
    prep_fit<<<1024, TPB, 0, stream>>>(fw0, fw1, fw2, F0t, F1t, F2t);
    embed_kernel<<<8192, TPB, 0, stream>>>(Ri, c0, w0, b1p, b2p, W1t, W2t, H);
    fit_kernel<<<512, TPB, 0, stream>>>(H, F0t, F1t, F2t, fb0p, fb1p, fb2p,
                                        fw3p, fb3c, (float*)d_out);
    etot_kernel<<<4, TPB, 0, stream>>>((const float*)d_out + 4, (float*)d_out);
}
